// MultiHeadAttention_63127429317070
// MI455X (gfx1250) — compile-verified, matrix operand order checked by reading, not run on hardware
//
#include <hip/hip_runtime.h>
#include <stdint.h>

#define B_  4
#define L_  2048
#define C_  1024
#define D_  1024
#define H_  16
#define DK_ 64

typedef __attribute__((ext_vector_type(16))) __bf16       v16bf;
typedef __attribute__((ext_vector_type(8)))  float        v8f;
typedef __attribute__((ext_vector_type(4)))  unsigned int v4u;

struct Frag32 { v4u lo, hi; };   // 32 bytes == v16bf

static __device__ __forceinline__ unsigned short f32_to_bf16(float f) {
  union { float f; unsigned int u; } c; c.f = f;
  unsigned int u = c.u;
  unsigned int r = u + 0x7FFFu + ((u >> 16) & 1u);  // round-to-nearest-even
  return (unsigned short)(r >> 16);
}

// Load a 16x32 bf16 fragment in the CDNA5 WMMA "A-like" layout from a
// row-major bf16 matrix (also used for B when the matrix is passed as B^T
// [N,K]).  Lane<16: row r0+lane, K k0+0..7 and k0+16..23; lane>=16: row
// r0+lane-16, K k0+8..15 and k0+24..31.  Two 16B loads per lane
// (global_load_b128 or ds_load_b128 depending on inferred address space).
static __device__ __forceinline__ v16bf load_frag_a(const unsigned short* p,
                                                    int ld, int r0, int k0) {
  int lane = (int)(threadIdx.x & 31u);
  int half = lane >> 4;
  const unsigned short* base = p + (size_t)(r0 + (lane & 15)) * (size_t)ld + k0 + half * 8;
  Frag32 f;
  f.lo = *(const v4u*)(base);        // 16 bytes: K +0..7  (or +8..15)
  f.hi = *(const v4u*)(base + 16);   // 16 bytes: K +16..23 (or +24..31)
  return __builtin_bit_cast(v16bf, f);
}

static __device__ __forceinline__ v8f wmma_bf16(v16bf a, v16bf b, v8f c) {
  // v_wmma_f32_16x16x32_bf16
  return __builtin_amdgcn_wmma_f32_16x16x32_bf16(false, a, false, b, (short)0, c, false, false);
}

// CDNA5 async copy: global -> LDS, 16 bytes per lane, tracked by ASYNCcnt.
// ldst is the low 32 bits of a generic pointer into __shared__ (== the LDS
// byte offset; hardware adds the workgroup LDS base).
static __device__ __forceinline__ void async_load_b128(const unsigned short* gsrc,
                                                       unsigned short* ldst) {
  unsigned lds = (unsigned)(uintptr_t)ldst;
  asm volatile("global_load_async_to_lds_b128 %0, %1, off"
               :: "v"(lds), "v"(gsrc) : "memory");
}
static __device__ __forceinline__ void wait_async0() {
  asm volatile("s_wait_asynccnt 0x0" ::: "memory");
}

// ---------------------------------------------------------------------------
// elementwise fp32 -> bf16 cast
__global__ void cast_bf16_kernel(const float* __restrict__ in,
                                 unsigned short* __restrict__ out, int n) {
  int i = blockIdx.x * blockDim.x + threadIdx.x;
  if (i < n) out[i] = f32_to_bf16(in[i]);
}

// W [K,N] fp32 (row-major)  ->  WT [N,K] bf16 (row-major)
__global__ void transpose_cast_kernel(const float* __restrict__ in,
                                      unsigned short* __restrict__ out,
                                      int K, int N) {
  int i = blockIdx.x * blockDim.x + threadIdx.x;
  if (i >= K * N) return;
  int k = i % K;
  int n = i / K;
  out[i] = f32_to_bf16(in[(size_t)k * N + n]);
}

// vp [B*L, D] fp32  ->  VT [B,H,DK,L] bf16   (V^T per head, K-major for WMMA B)
__global__ void make_vt_kernel(const float* __restrict__ vp,
                               unsigned short* __restrict__ VT) {
  size_t i = (size_t)blockIdx.x * blockDim.x + threadIdx.x;  // B*H*DK*L
  int l = (int)(i % L_);
  size_t t = i / L_;
  int d = (int)(t % DK_); t /= DK_;
  int h = (int)(t % H_);
  int b = (int)(t / H_);
  VT[i] = f32_to_bf16(vp[((size_t)b * L_ + l) * D_ + h * DK_ + d]);
}

// ---------------------------------------------------------------------------
// bf16 WMMA GEMM:  C[M,N] = A[M,K] @ B[K,N]  with B supplied transposed
// (BT [N,K] row-major).  Wave tile 32x64, block = 4 waves => block tile
// 128x64.  All 4 waves share one 64x32 B tile, double-buffered in LDS via
// GLOBAL_LOAD_ASYNC_TO_LDS_B128 (ASYNCcnt) and read back with ds_load_b128.
__global__ __launch_bounds__(128)
void gemm_bf16_kernel(const unsigned short* __restrict__ A,
                      const unsigned short* __restrict__ BT,
                      float* __restrict__ Cf,
                      unsigned short* __restrict__ Cb,
                      const float* __restrict__ residual,
                      const float* __restrict__ bias,
                      int M, int N, int K) {
  __shared__ __align__(16) unsigned short bbuf[2][64 * 32];  // 2 x 4KB

  const int tid  = (int)threadIdx.x;
  const int lane = tid & 31;
  const int w    = tid >> 5;
  const int half = lane >> 4;
  const int m0 = blockIdx.x * 128 + w * 32;
  const int n0 = blockIdx.y * 64;

  // Stage BT rows [n0..n0+63] x cols [kk..kk+31] into bbuf[s]:
  // 64 rows x 32 bf16 = 256 chunks of 16B; exactly 2 chunks per thread,
  // issued straight-line (no divergent branch around the inline asm).
  const int srow = tid >> 2;          // 0..31
  const int scol = (tid & 3) * 8;     // 0,8,16,24
  auto stage_b = [&](int s, int kk) {
    const unsigned short* g0 = BT + (size_t)(n0 + srow) * K + (kk + scol);
    async_load_b128(g0, &bbuf[s][srow * 32 + scol]);
    const unsigned short* g1 = BT + (size_t)(n0 + srow + 32) * K + (kk + scol);
    async_load_b128(g1, &bbuf[s][(srow + 32) * 32 + scol]);
  };

  const v8f vzero = {0.f, 0.f, 0.f, 0.f, 0.f, 0.f, 0.f, 0.f};
  v8f acc[2][4];
#pragma unroll
  for (int i = 0; i < 2; ++i)
#pragma unroll
    for (int j = 0; j < 4; ++j) acc[i][j] = vzero;

  stage_b(0, 0);
  for (int kk = 0; kk < K; kk += 32) {
    const int s = (kk >> 5) & 1;
    wait_async0();        // our async loads into bbuf[s] have landed
    __syncthreads();      // everyone's have; also fences reads of bbuf[s^1]
    if (kk + 32 < K) stage_b(s ^ 1, kk + 32);  // overlap next tile with compute

    v16bf a0 = load_frag_a(A, K, m0,      kk);
    v16bf a1 = load_frag_a(A, K, m0 + 16, kk);
    v16bf b0 = load_frag_a(bbuf[s], 32, 0,  0);
    v16bf b1 = load_frag_a(bbuf[s], 32, 16, 0);
    v16bf b2 = load_frag_a(bbuf[s], 32, 32, 0);
    v16bf b3 = load_frag_a(bbuf[s], 32, 48, 0);
    acc[0][0] = wmma_bf16(a0, b0, acc[0][0]);
    acc[0][1] = wmma_bf16(a0, b1, acc[0][1]);
    acc[0][2] = wmma_bf16(a0, b2, acc[0][2]);
    acc[0][3] = wmma_bf16(a0, b3, acc[0][3]);
    acc[1][0] = wmma_bf16(a1, b0, acc[1][0]);
    acc[1][1] = wmma_bf16(a1, b1, acc[1][1]);
    acc[1][2] = wmma_bf16(a1, b2, acc[1][2]);
    acc[1][3] = wmma_bf16(a1, b3, acc[1][3]);
  }

#pragma unroll
  for (int i = 0; i < 2; ++i)
#pragma unroll
    for (int j = 0; j < 4; ++j)
#pragma unroll
      for (int r = 0; r < 8; ++r) {
        int row = m0 + i * 16 + r + 8 * half;
        int col = n0 + j * 16 + (lane & 15);
        float vv = acc[i][j][r];
        if (residual) vv += residual[(size_t)row * N + col];
        if (bias)     vv += bias[col];
        if (Cf) Cf[(size_t)row * N + col] = vv;
        if (Cb) Cb[(size_t)row * N + col] = f32_to_bf16(vv);
      }
}

// ---------------------------------------------------------------------------
// Flash attention: one wave per 16 query rows of one (b,h).  Online softmax,
// S computed 16x32 per step (4 WMMAs), P bounced through 1KB LDS into the
// A-fragment layout, PV accumulated into a 16x64 f32 accumulator (4 WMMAs).
__global__ __launch_bounds__(128)
void flash_attn_kernel(const unsigned short* __restrict__ Qp,   // [B*L, D] bf16
                       const unsigned short* __restrict__ Kp,   // [B*L, D] bf16
                       const unsigned short* __restrict__ VT,   // [B,H,DK,L] bf16
                       unsigned short* __restrict__ ctx) {      // [B*L, D] bf16
  __shared__ __align__(16) unsigned short pbuf[4][16 * 32];
  const int lane = (int)(threadIdx.x & 31u);
  const int w    = (int)(threadIdx.x >> 5u);
  const int half = lane >> 4;

  const int gid = blockIdx.x * 4 + w;       // 0 .. B*H*(L/16)-1
  const int qt = gid & 127;                 // L/16 = 128 tiles
  const int h  = (gid >> 7) & 15;
  const int b  = gid >> 11;
  const int q0 = qt * 16;

  const unsigned short* Qb = Qp + (size_t)b * L_ * D_ + h * DK_;
  const unsigned short* Kb = Kp + (size_t)b * L_ * D_ + h * DK_;
  const unsigned short* Vb = VT + (size_t)(b * H_ + h) * DK_ * L_;
  unsigned short* pb = pbuf[w];

  const v16bf aq0 = load_frag_a(Qb, D_, q0, 0);
  const v16bf aq1 = load_frag_a(Qb, D_, q0, 32);

  const v8f vzero = {0.f, 0.f, 0.f, 0.f, 0.f, 0.f, 0.f, 0.f};
  v8f o[4];
  float mrow[8], lrow[8];
#pragma unroll
  for (int t = 0; t < 4; ++t) o[t] = vzero;
#pragma unroll
  for (int r = 0; r < 8; ++r) { mrow[r] = -1e30f; lrow[r] = 0.f; }

  const float scale = 0.125f;  // 1/sqrt(64)

  for (int kk = 0; kk < L_; kk += 32) {
    v8f s0 = vzero, s1 = vzero;
    s0 = wmma_bf16(aq0, load_frag_a(Kb, D_, kk,      0),  s0);
    s0 = wmma_bf16(aq1, load_frag_a(Kb, D_, kk,      32), s0);
    s1 = wmma_bf16(aq0, load_frag_a(Kb, D_, kk + 16, 0),  s1);
    s1 = wmma_bf16(aq1, load_frag_a(Kb, D_, kk + 16, 32), s1);

    float alpha[8];
#pragma unroll
    for (int r = 0; r < 8; ++r) {
      float a0 = s0[r] * scale;
      float a1 = s1[r] * scale;
      // row max over 16 lanes of this half-wave (row m = r + 8*half)
      float mx = fmaxf(a0, a1);
#pragma unroll
      for (int off = 1; off < 16; off <<= 1)
        mx = fmaxf(mx, __shfl_xor(mx, off, 32));
      float mn = fmaxf(mrow[r], mx);
      alpha[r] = __expf(mrow[r] - mn);
      mrow[r] = mn;
      float p0 = __expf(a0 - mn);
      float p1 = __expf(a1 - mn);
      float rs = p0 + p1;
#pragma unroll
      for (int off = 1; off < 16; off <<= 1)
        rs += __shfl_xor(rs, off, 32);
      lrow[r] = lrow[r] * alpha[r] + rs;
      // stash P tile (16x32) into LDS, C-layout -> row-major
      pb[(r + 8 * half) * 32 + (lane & 15)]      = f32_to_bf16(p0);
      pb[(r + 8 * half) * 32 + 16 + (lane & 15)] = f32_to_bf16(p1);
    }
    // rescale O by alpha (per row)
#pragma unroll
    for (int t = 0; t < 4; ++t)
#pragma unroll
      for (int r = 0; r < 8; ++r) o[t][r] *= alpha[r];

    // reload P in A-fragment layout (same-wave LDS ops are in order)
    v16bf ap = load_frag_a(pb, 32, 0, 0);
#pragma unroll
    for (int t = 0; t < 4; ++t)
      o[t] = wmma_bf16(ap, load_frag_a(Vb, L_, t * 16, kk), o[t]);
  }

#pragma unroll
  for (int r = 0; r < 8; ++r) lrow[r] = 1.0f / lrow[r];
#pragma unroll
  for (int t = 0; t < 4; ++t)
#pragma unroll
    for (int r = 0; r < 8; ++r) {
      size_t row = (size_t)b * L_ + q0 + r + 8 * half;
      int col = h * DK_ + t * 16 + (lane & 15);
      ctx[row * D_ + col] = f32_to_bf16(o[t][r] * lrow[r]);
    }
}

// ---------------------------------------------------------------------------
// LayerNorm over D=1024 per row, eps=1e-6, bf16 output for the final GEMM.
__global__ __launch_bounds__(256)
void layernorm_kernel(const float* __restrict__ x,
                      const float* __restrict__ g,
                      const float* __restrict__ beta,
                      unsigned short* __restrict__ out) {
  __shared__ float red[256];
  const int row = blockIdx.x;
  const float* xr = x + (size_t)row * D_;

  float s = 0.f;
  for (int i = threadIdx.x; i < D_; i += 256) s += xr[i];
  red[threadIdx.x] = s;
  __syncthreads();
  for (int off = 128; off > 0; off >>= 1) {
    if ((int)threadIdx.x < off) red[threadIdx.x] += red[threadIdx.x + off];
    __syncthreads();
  }
  float mu = red[0] * (1.0f / D_);
  __syncthreads();

  float v = 0.f;
  for (int i = threadIdx.x; i < D_; i += 256) {
    float d = xr[i] - mu;
    v += d * d;
  }
  red[threadIdx.x] = v;
  __syncthreads();
  for (int off = 128; off > 0; off >>= 1) {
    if ((int)threadIdx.x < off) red[threadIdx.x] += red[threadIdx.x + off];
    __syncthreads();
  }
  float rstd = rsqrtf(red[0] * (1.0f / D_) + 1e-6f);
  __syncthreads();

  for (int i = threadIdx.x; i < D_; i += 256)
    out[(size_t)row * D_ + i] = f32_to_bf16((xr[i] - mu) * rstd * g[i] + beta[i]);
}

// ---------------------------------------------------------------------------
extern "C" void kernel_launch(void* const* d_in, const int* in_sizes, int n_in,
                              void* d_out, int out_size, void* d_ws, size_t ws_size,
                              hipStream_t stream) {
  const float* q    = (const float*)d_in[0];
  const float* k    = (const float*)d_in[1];
  const float* v    = (const float*)d_in[2];
  const float* Wq   = (const float*)d_in[3];
  const float* Wk   = (const float*)d_in[4];
  const float* Wv   = (const float*)d_in[5];
  const float* Wfc  = (const float*)d_in[6];
  const float* ln_g = (const float*)d_in[7];
  const float* ln_b = (const float*)d_in[8];
  const float* Wout = (const float*)d_in[9];
  const float* bout = (const float*)d_in[10];
  float* out = (float*)d_out;

  const size_t nBLC = (size_t)B_ * L_ * C_;  // 8388608 (== B*L*D)
  const int M = B_ * L_;                      // 8192

  char* ws = (char*)d_ws;
  size_t cur = 0;
  auto alloc = [&](size_t bytes) -> void* {
    void* p = ws + cur;
    cur = (cur + bytes + 255) & ~(size_t)255;
    return p;
  };

  unsigned short* qb    = (unsigned short*)alloc(nBLC * 2);
  unsigned short* kb    = (unsigned short*)alloc(nBLC * 2);
  unsigned short* vb    = (unsigned short*)alloc(nBLC * 2);
  unsigned short* WqT   = (unsigned short*)alloc((size_t)C_ * D_ * 2);
  unsigned short* WkT   = (unsigned short*)alloc((size_t)C_ * D_ * 2);
  unsigned short* WvT   = (unsigned short*)alloc((size_t)C_ * D_ * 2);
  unsigned short* WfcT  = (unsigned short*)alloc((size_t)D_ * D_ * 2);
  unsigned short* WoutT = (unsigned short*)alloc((size_t)D_ * C_ * 2);
  unsigned short* Qp    = (unsigned short*)alloc(nBLC * 2);
  unsigned short* Kp    = (unsigned short*)alloc(nBLC * 2);
  float*          vp    = (float*)alloc(nBLC * 4);
  unsigned short* VT    = (unsigned short*)alloc(nBLC * 2);
  unsigned short* ctx   = (unsigned short*)alloc(nBLC * 2);
  float*          fc    = (float*)alloc(nBLC * 4);
  unsigned short* lnb   = (unsigned short*)alloc(nBLC * 2);

  // 1) fp32 -> bf16 casts of activations
  int nElems = (int)nBLC;
  cast_bf16_kernel<<<nElems / 256, 256, 0, stream>>>(q, qb, nElems);
  cast_bf16_kernel<<<nElems / 256, 256, 0, stream>>>(k, kb, nElems);
  cast_bf16_kernel<<<nElems / 256, 256, 0, stream>>>(v, vb, nElems);

  // 2) weights -> bf16 [N,K]
  transpose_cast_kernel<<<(C_ * D_) / 256, 256, 0, stream>>>(Wq,   WqT,   C_, D_);
  transpose_cast_kernel<<<(C_ * D_) / 256, 256, 0, stream>>>(Wk,   WkT,   C_, D_);
  transpose_cast_kernel<<<(C_ * D_) / 256, 256, 0, stream>>>(Wv,   WvT,   C_, D_);
  transpose_cast_kernel<<<(D_ * D_) / 256, 256, 0, stream>>>(Wfc,  WfcT,  D_, D_);
  transpose_cast_kernel<<<(D_ * C_) / 256, 256, 0, stream>>>(Wout, WoutT, D_, C_);

  // 3) projections (WMMA)
  dim3 ggrid(M / 128, D_ / 64);
  gemm_bf16_kernel<<<ggrid, 128, 0, stream>>>(qb, WqT, nullptr, Qp, nullptr, nullptr, M, D_, C_);
  gemm_bf16_kernel<<<ggrid, 128, 0, stream>>>(kb, WkT, nullptr, Kp, nullptr, nullptr, M, D_, C_);
  gemm_bf16_kernel<<<ggrid, 128, 0, stream>>>(vb, WvT, vp, nullptr, nullptr, nullptr, M, D_, C_);

  // 4) per-head V^T for the PV matmul
  make_vt_kernel<<<(int)(nBLC / 256), 256, 0, stream>>>(vp, VT);

  // 5) flash attention (WMMA)
  flash_attn_kernel<<<(B_ * H_ * (L_ / 16)) / 4, 128, 0, stream>>>(Qp, Kp, VT, ctx);

  // 6) FC + residual (WMMA, f32 out)
  gemm_bf16_kernel<<<ggrid, 128, 0, stream>>>(ctx, WfcT, fc, nullptr, vp, nullptr, M, D_, D_);

  // 7) LayerNorm -> bf16
  layernorm_kernel<<<M, 256, 0, stream>>>(fc, ln_g, ln_b, lnb);

  // 8) output projection + bias (WMMA, f32 out)
  dim3 ogrid(M / 128, C_ / 64);
  gemm_bf16_kernel<<<ogrid, 128, 0, stream>>>(lnb, WoutT, out, nullptr, nullptr, bout, M, C_, D_);
}
